// MoD_3513283248419
// MI455X (gfx1250) — matrix operand order repaired and checked by reading, not compiled, not run on hardware
//
#include <hip/hip_runtime.h>
#include <math.h>

typedef __attribute__((ext_vector_type(16))) __bf16 v16bf;
typedef __attribute__((ext_vector_type(8)))  float  v8f;

#define WMMA_BF16(a,b,c) __builtin_amdgcn_wmma_f32_16x16x32_bf16(false,(a),false,(b),(short)0,(c),false,false)

#define DMODEL 1024
#define SEQ    4096
#define KSEL   2048
#define NHEAD  16
#define DHEAD  64
#define DFF    4096
#define NBATCH 4

// ----------------------------------------------------------------------------
// Utility kernels
// ----------------------------------------------------------------------------
__global__ __launch_bounds__(256) void k_copy(const float* __restrict__ s,
                                              float* __restrict__ d, int n) {
  int i = blockIdx.x * 256 + threadIdx.x;
  if (i < n) d[i] = s[i];
}

// Transpose + convert: s is K x N (f32, row-major) -> d is N x K (bf16, row-major)
__global__ __launch_bounds__(256) void k_f2bf_t(const float* __restrict__ s,
                                                __bf16* __restrict__ d, int K, int N) {
  __shared__ float tile[32][33];
  int n0 = blockIdx.x * 32, k0 = blockIdx.y * 32;
  int tx = threadIdx.x & 31, ty = threadIdx.x >> 5;  // 32 x 8
#pragma unroll
  for (int i = 0; i < 32; i += 8)
    tile[ty + i][tx] = s[(size_t)(k0 + ty + i) * N + n0 + tx];
  __syncthreads();
#pragma unroll
  for (int i = 0; i < 32; i += 8)
    d[(size_t)(n0 + ty + i) * K + k0 + tx] = (__bf16)tile[tx][ty + i];
}

// Router logits: one wave32 per token row, shuffle reduction.
__global__ __launch_bounds__(256) void k_router(const float* __restrict__ x,
                                                const float* __restrict__ w,
                                                float* __restrict__ logits) {
  int wid  = (blockIdx.x * 256 + threadIdx.x) >> 5;
  int lane = threadIdx.x & 31;
  const float* row = x + (size_t)wid * DMODEL;
  float acc = 0.f;
  for (int i = lane; i < DMODEL; i += 32) acc += row[i] * w[i];
#pragma unroll
  for (int off = 16; off > 0; off >>= 1) acc += __shfl_down(acc, off, 32);
  if (lane == 0) logits[wid] = acc;
}

// ----------------------------------------------------------------------------
// Top-K selection (per batch, one 1024-thread block)
// ----------------------------------------------------------------------------
__device__ __forceinline__ unsigned blk_reduce_add_u(unsigned v, unsigned* red, int t) {
  red[t] = v; __syncthreads();
  for (int off = 512; off > 0; off >>= 1) { if (t < off) red[t] += red[t + off]; __syncthreads(); }
  unsigned r = red[0]; __syncthreads(); return r;
}
__device__ __forceinline__ float blk_reduce_add_f(float v, float* red, int t) {
  red[t] = v; __syncthreads();
  for (int off = 512; off > 0; off >>= 1) { if (t < off) red[t] += red[t + off]; __syncthreads(); }
  float r = red[0]; __syncthreads(); return r;
}
__device__ __forceinline__ float blk_reduce_max_f(float v, float* red, int t) {
  red[t] = v; __syncthreads();
  for (int off = 512; off > 0; off >>= 1) { if (t < off) red[t] = fmaxf(red[t], red[t + off]); __syncthreads(); }
  float r = red[0]; __syncthreads(); return r;
}
__device__ __forceinline__ unsigned blk_scan_incl_u(unsigned v, unsigned* red, int t) {
  red[t] = v; __syncthreads();
  for (int off = 1; off < 1024; off <<= 1) {
    unsigned a = (t >= off) ? red[t - off] : 0u;
    __syncthreads();
    red[t] += a;
    __syncthreads();
  }
  unsigned r = red[t]; __syncthreads(); return r;
}

__global__ __launch_bounds__(1024) void k_topk(const float* __restrict__ logits, int b,
                                               int* __restrict__ sel, float* __restrict__ rw) {
  __shared__ unsigned keysS[SEQ];
  __shared__ float    lgS[SEQ];
  __shared__ unsigned redU[1024];
  __shared__ float    redF[1024];
  __shared__ float    seS[KSEL];
  int t = threadIdx.x;
  const float* L = logits + (size_t)b * SEQ;
  for (int i = t; i < SEQ; i += 1024) {
    float f = L[i];
    lgS[i] = f;
    unsigned u = __float_as_uint(f);
    keysS[i] = (u & 0x80000000u) ? ~u : (u | 0x80000000u);  // order-preserving
  }
  __syncthreads();

  unsigned thresh = 0;
  for (int bit = 31; bit >= 0; --bit) {
    unsigned cand = thresh | (1u << bit);
    unsigned c = 0;
    for (int i = t; i < SEQ; i += 1024) c += (keysS[i] >= cand) ? 1u : 0u;
    unsigned total = blk_reduce_add_u(c, redU, t);
    if (total >= (unsigned)KSEL) thresh = cand;
  }
  unsigned cg = 0;
  for (int i = t; i < SEQ; i += 1024) cg += (keysS[i] > thresh) ? 1u : 0u;
  unsigned G = blk_reduce_add_u(cg, redU, t);
  unsigned needEq = (unsigned)KSEL - G;

  unsigned eqc = 0;
#pragma unroll
  for (int k = 0; k < 4; k++) eqc += (keysS[t * 4 + k] == thresh) ? 1u : 0u;
  unsigned eqExcl = blk_scan_incl_u(eqc, redU, t) - eqc;

  unsigned f4[4]; unsigned fc = 0; unsigned eqRun = eqExcl;
#pragma unroll
  for (int k = 0; k < 4; k++) {
    int i = t * 4 + k;
    unsigned kk = keysS[i];
    unsigned flg = (kk > thresh) ? 1u : ((kk == thresh && eqRun < needEq) ? 1u : 0u);
    if (kk == thresh) eqRun++;
    f4[k] = flg; fc += flg;
  }
  unsigned pos = blk_scan_incl_u(fc, redU, t) - fc;

  float mloc = -3.4e38f;
  for (int i = t; i < SEQ; i += 1024) mloc = fmaxf(mloc, lgS[i]);
  float m = blk_reduce_max_f(mloc, redF, t);

  float esum = 0.f; unsigned p = pos;
#pragma unroll
  for (int k = 0; k < 4; k++) {
    if (f4[k]) {
      int i = t * 4 + k;
      float e = __expf(lgS[i] - m);
      seS[p] = e;
      sel[b * KSEL + p] = i;
      esum += e; p++;
    }
  }
  float sum = blk_reduce_add_f(esum, redF, t);
  float inv = 1.f / sum;
  for (int j = t; j < KSEL; j += 1024) rw[b * KSEL + j] = seS[j] * inv;
}

// ----------------------------------------------------------------------------
// Gather + LayerNorm1 -> fx (f32 residual) and h (bf16)
// ----------------------------------------------------------------------------
__global__ __launch_bounds__(256) void k_gather_ln1(const float* __restrict__ x,
                                                    const int* __restrict__ sel,
                                                    const float* __restrict__ g,
                                                    const float* __restrict__ be,
                                                    float* __restrict__ fx,
                                                    __bf16* __restrict__ h, int b) {
  __shared__ float red[256];
  __shared__ float sMean, sRstd;
  int j = blockIdx.x, t = threadIdx.x;
  int row = sel[b * KSEL + j];
  const float* src = x + ((size_t)b * SEQ + row) * DMODEL;
  float v[4]; float s = 0.f;
#pragma unroll
  for (int k = 0; k < 4; k++) { v[k] = src[t + 256 * k]; s += v[k]; }
  red[t] = s; __syncthreads();
  for (int off = 128; off > 0; off >>= 1) { if (t < off) red[t] += red[t + off]; __syncthreads(); }
  if (t == 0) sMean = red[0] * (1.f / DMODEL);
  __syncthreads();
  float mean = sMean; float s2 = 0.f;
#pragma unroll
  for (int k = 0; k < 4; k++) { float d0 = v[k] - mean; s2 += d0 * d0; }
  red[t] = s2; __syncthreads();
  for (int off = 128; off > 0; off >>= 1) { if (t < off) red[t] += red[t + off]; __syncthreads(); }
  if (t == 0) sRstd = rsqrtf(red[0] * (1.f / DMODEL) + 1e-5f);
  __syncthreads();
  float rstd = sRstd;
#pragma unroll
  for (int k = 0; k < 4; k++) {
    int c = t + 256 * k;
    fx[(size_t)j * DMODEL + c] = v[k];
    h[(size_t)j * DMODEL + c] = (__bf16)((v[k] - mean) * rstd * g[c] + be[c]);
  }
}

// LayerNorm2: x1 (f32) -> h2 (bf16)
__global__ __launch_bounds__(256) void k_ln2(const float* __restrict__ x1,
                                             const float* __restrict__ g,
                                             const float* __restrict__ be,
                                             __bf16* __restrict__ h2) {
  __shared__ float red[256];
  __shared__ float sMean, sRstd;
  int j = blockIdx.x, t = threadIdx.x;
  const float* src = x1 + (size_t)j * DMODEL;
  float v[4]; float s = 0.f;
#pragma unroll
  for (int k = 0; k < 4; k++) { v[k] = src[t + 256 * k]; s += v[k]; }
  red[t] = s; __syncthreads();
  for (int off = 128; off > 0; off >>= 1) { if (t < off) red[t] += red[t + off]; __syncthreads(); }
  if (t == 0) sMean = red[0] * (1.f / DMODEL);
  __syncthreads();
  float mean = sMean; float s2 = 0.f;
#pragma unroll
  for (int k = 0; k < 4; k++) { float d0 = v[k] - mean; s2 += d0 * d0; }
  red[t] = s2; __syncthreads();
  for (int off = 128; off > 0; off >>= 1) { if (t < off) red[t] += red[t + off]; __syncthreads(); }
  if (t == 0) sRstd = rsqrtf(red[0] * (1.f / DMODEL) + 1e-5f);
  __syncthreads();
  float rstd = sRstd;
#pragma unroll
  for (int k = 0; k < 4; k++) {
    int c = t + 256 * k;
    h2[(size_t)j * DMODEL + c] = (__bf16)((v[k] - mean) * rstd * g[c] + be[c]);
  }
}

// ----------------------------------------------------------------------------
// bf16 WMMA GEMM, direct-from-L2 streaming (no LDS):
//   C[M,N] = A[M,K] @ BT[N,K]^T, f32 accumulate.
// Block tile 128x128, 8 waves of 32x64, K-step 32.
// Per-lane fragment reads are contiguous 16B runs -> global_load_b128 clauses.
// MODE 0: f32 out   MODE 1: bf16 out   MODE 2: f32 out += Res   MODE 3: GELU->bf16
// ----------------------------------------------------------------------------
template <int MODE>
__global__ __launch_bounds__(256) void k_gemm(const __bf16* __restrict__ A,
                                              const __bf16* __restrict__ BT,
                                              int M, int N, int Kd,
                                              float* __restrict__ Cf,
                                              __bf16* __restrict__ Cb,
                                              const float* __restrict__ Res) {
  int t = threadIdx.x;
  int lane = t & 31, wave = t >> 5;
  int bm = blockIdx.y, bn = blockIdx.x;
  int wm = wave >> 1, wn = wave & 1;
  bool hi = lane >= 16; int l15 = lane & 15;

  const __bf16* Arow = A  + (size_t)(bm * 128 + wm * 32 + l15) * Kd;
  const __bf16* Brow = BT + (size_t)(bn * 128 + wn * 64 + l15) * Kd;

  v8f c[2][4];
#pragma unroll
  for (int i = 0; i < 2; i++)
#pragma unroll
    for (int j = 0; j < 4; j++) c[i][j] = (v8f){};

#pragma unroll 2
  for (int k0 = 0; k0 < Kd; k0 += 32) {
    v16bf a[2], bfr[4];
#pragma unroll
    for (int fm = 0; fm < 2; fm++)
#pragma unroll
      for (int j = 0; j < 16; j++) {
        int kk = j + (hi ? 16 : 0);
        a[fm][j] = Arow[(size_t)fm * 16 * Kd + k0 + kk];
      }
#pragma unroll
    for (int fn = 0; fn < 4; fn++)
#pragma unroll
      for (int j = 0; j < 16; j++) {
        int kk = j + (hi ? 16 : 0);
        bfr[fn][j] = Brow[(size_t)fn * 16 * Kd + k0 + kk];
      }
#pragma unroll
    for (int fm = 0; fm < 2; fm++)
#pragma unroll
      for (int fn = 0; fn < 4; fn++) c[fm][fn] = WMMA_BF16(a[fm], bfr[fn], c[fm][fn]);
  }

#pragma unroll
  for (int fm = 0; fm < 2; fm++)
#pragma unroll
    for (int fn = 0; fn < 4; fn++)
#pragma unroll
      for (int r = 0; r < 8; r++) {
        int row = bm * 128 + wm * 32 + fm * 16 + r + (hi ? 8 : 0);
        int col = bn * 128 + wn * 64 + fn * 16 + l15;
        size_t off = (size_t)row * N + col;
        float v = c[fm][fn][r];
        if (MODE == 0) Cf[off] = v;
        if (MODE == 1) Cb[off] = (__bf16)v;
        if (MODE == 2) Cf[off] = v + Res[off];
        if (MODE == 3) {
          float gl = 0.5f * v * (1.f + tanhf(0.7978845608f * (v + 0.044715f * v * v * v)));
          Cb[off] = (__bf16)gl;
        }
      }
}

// ----------------------------------------------------------------------------
// Flash attention: qkv (bf16, row stride 3072: [q|k|v] x 16 heads x 64)
// One wave per 16 q-rows; seq = KSEL keys in steps of 32.
// ----------------------------------------------------------------------------
__global__ __launch_bounds__(256) void k_flash(const __bf16* __restrict__ qkv,
                                               __bf16* __restrict__ o) {
  __shared__ __bf16 pbuf[8][16 * 32];
  int t = threadIdx.x;
  int lane = t & 31, wave = t >> 5;
  int head = blockIdx.x >> 4;
  int qblk = blockIdx.x & 15;
  int qbase = qblk * 128 + wave * 16;
  bool hi = lane >= 16; int l15 = lane & 15;
  const int LD = 3 * DMODEL;
  const float scale = 0.125f;  // 1/sqrt(64)

  v16bf aq[2];
#pragma unroll
  for (int f = 0; f < 2; f++)
#pragma unroll
    for (int j = 0; j < 16; j++) {
      int dk = f * 32 + j + (hi ? 16 : 0);
      aq[f][j] = qkv[(size_t)(qbase + l15) * LD + head * DHEAD + dk];
    }

  v8f acc[4];
#pragma unroll
  for (int nb = 0; nb < 4; nb++) acc[nb] = (v8f){};
  float mrow[8], lrow[8];
#pragma unroll
  for (int r = 0; r < 8; r++) { mrow[r] = -3.4e38f; lrow[r] = 0.f; }

  for (int kt = 0; kt < KSEL; kt += 32) {
    v8f S[2]; S[0] = (v8f){}; S[1] = (v8f){};
#pragma unroll
    for (int sc = 0; sc < 2; sc++) {
#pragma unroll
      for (int f = 0; f < 2; f++) {
        v16bf bk;
#pragma unroll
        for (int j = 0; j < 16; j++) {
          int kk = j + (hi ? 16 : 0);
          bk[j] = qkv[(size_t)(kt + sc * 16 + l15) * LD + DMODEL + head * DHEAD + f * 32 + kk];
        }
        S[sc] = WMMA_BF16(aq[f], bk, S[sc]);
      }
    }
    float rescale[8];
#pragma unroll
    for (int r = 0; r < 8; r++) {
      float s0 = S[0][r] * scale, s1 = S[1][r] * scale;
      float mx = fmaxf(s0, s1);
#pragma unroll
      for (int off = 8; off > 0; off >>= 1) mx = fmaxf(mx, __shfl_xor(mx, off, 32));
      float mnew = fmaxf(mrow[r], mx);
      rescale[r] = __expf(mrow[r] - mnew);
      float p0 = __expf(s0 - mnew), p1 = __expf(s1 - mnew);
      S[0][r] = p0; S[1][r] = p1;
      float ps = p0 + p1;
#pragma unroll
      for (int off = 8; off > 0; off >>= 1) ps += __shfl_xor(ps, off, 32);
      lrow[r] = lrow[r] * rescale[r] + ps;
      mrow[r] = mnew;
    }
#pragma unroll
    for (int nb = 0; nb < 4; nb++)
#pragma unroll
      for (int r = 0; r < 8; r++) acc[nb][r] *= rescale[r];

    // P: C-layout -> A-layout through per-wave LDS (bf16)
#pragma unroll
    for (int sc = 0; sc < 2; sc++)
#pragma unroll
      for (int r = 0; r < 8; r++)
        pbuf[wave][(r + (hi ? 8 : 0)) * 32 + sc * 16 + l15] = (__bf16)S[sc][r];
    asm volatile("s_wait_dscnt 0" ::: "memory");
    v16bf ap;
#pragma unroll
    for (int j = 0; j < 16; j++) {
      int kk = j + (hi ? 16 : 0);
      ap[j] = pbuf[wave][l15 * 32 + kk];
    }
#pragma unroll
    for (int nb = 0; nb < 4; nb++) {
      v16bf bv;
#pragma unroll
      for (int j = 0; j < 16; j++) {
        int kk = j + (hi ? 16 : 0);
        bv[j] = qkv[(size_t)(kt + kk) * LD + 2 * DMODEL + head * DHEAD + nb * 16 + l15];
      }
      acc[nb] = WMMA_BF16(ap, bv, acc[nb]);
    }
  }

#pragma unroll
  for (int nb = 0; nb < 4; nb++)
#pragma unroll
    for (int r = 0; r < 8; r++) {
      int row = qbase + r + (hi ? 8 : 0);
      int col = head * DHEAD + nb * 16 + l15;
      o[(size_t)row * DMODEL + col] = (__bf16)(acc[nb][r] / lrow[r]);
    }
}

// Scatter: out[b, sel[j], :] += rw[j] * xo[j, :]
__global__ __launch_bounds__(256) void k_scatter(float* __restrict__ out,
                                                 const int* __restrict__ sel,
                                                 const float* __restrict__ rw,
                                                 const float* __restrict__ xo, int b) {
  int j = blockIdx.x, t = threadIdx.x;
  int row = sel[b * KSEL + j];
  float w = rw[b * KSEL + j];
  float* dst = out + ((size_t)b * SEQ + row) * DMODEL;
  const float* src = xo + (size_t)j * DMODEL;
#pragma unroll
  for (int k = 0; k < 4; k++) { int c = t + 256 * k; dst[c] += w * src[c]; }
}

// ----------------------------------------------------------------------------
// Host launch
// ----------------------------------------------------------------------------
static inline char* carve(char*& p, size_t bytes) {
  char* r = p;
  p += (bytes + 255) & ~(size_t)255;
  return r;
}

extern "C" void kernel_launch(void* const* d_in, const int* in_sizes, int n_in,
                              void* d_out, int out_size, void* d_ws, size_t ws_size,
                              hipStream_t stream) {
  const float* x        = (const float*)d_in[0];
  const float* router_w = (const float*)d_in[1];
  const float* ln1_g    = (const float*)d_in[2];
  const float* ln1_b    = (const float*)d_in[3];
  const float* ln2_g    = (const float*)d_in[4];
  const float* ln2_b    = (const float*)d_in[5];
  const float* wqkv     = (const float*)d_in[6];
  const float* wo       = (const float*)d_in[7];
  const float* w1       = (const float*)d_in[8];
  const float* w2       = (const float*)d_in[9];
  float* out = (float*)d_out;

  char* p = (char*)d_ws;
  float*  logits  = (float*)carve(p, (size_t)NBATCH * SEQ * 4);
  int*    sel     = (int*)  carve(p, (size_t)NBATCH * KSEL * 4);
  float*  rw      = (float*)carve(p, (size_t)NBATCH * KSEL * 4);
  __bf16* wqkv_t  = (__bf16*)carve(p, (size_t)DMODEL * 3 * DMODEL * 2);  // (3d) x d
  __bf16* wo_t    = (__bf16*)carve(p, (size_t)DMODEL * DMODEL * 2);      // d x d
  __bf16* w1_t    = (__bf16*)carve(p, (size_t)DMODEL * DFF * 2);         // dff x d
  __bf16* w2_t    = (__bf16*)carve(p, (size_t)DFF * DMODEL * 2);         // d x dff
  float*  fx      = (float*)carve(p, (size_t)KSEL * DMODEL * 4);
  __bf16* h_bf    = (__bf16*)carve(p, (size_t)KSEL * DMODEL * 2);
  __bf16* qkv_bf  = (__bf16*)carve(p, (size_t)KSEL * 3 * DMODEL * 2);
  __bf16* o_bf    = (__bf16*)carve(p, (size_t)KSEL * DMODEL * 2);
  float*  x1      = (float*)carve(p, (size_t)KSEL * DMODEL * 4);
  __bf16* h2_bf   = (__bf16*)carve(p, (size_t)KSEL * DMODEL * 2);
  __bf16* g_bf    = (__bf16*)carve(p, (size_t)KSEL * DFF * 2);
  float*  xo      = (float*)carve(p, (size_t)KSEL * DMODEL * 4);

  // weights -> bf16, transposed to N x K for K-contiguous fragment loads
  k_f2bf_t<<<dim3(3 * DMODEL / 32, DMODEL / 32), 256, 0, stream>>>(wqkv, wqkv_t, DMODEL, 3 * DMODEL);
  k_f2bf_t<<<dim3(DMODEL / 32, DMODEL / 32), 256, 0, stream>>>(wo, wo_t, DMODEL, DMODEL);
  k_f2bf_t<<<dim3(DFF / 32, DMODEL / 32), 256, 0, stream>>>(w1, w1_t, DMODEL, DFF);
  k_f2bf_t<<<dim3(DMODEL / 32, DFF / 32), 256, 0, stream>>>(w2, w2_t, DFF, DMODEL);

  // out = x
  int ntot = NBATCH * SEQ * DMODEL;
  k_copy<<<(ntot + 255) / 256, 256, 0, stream>>>(x, out, ntot);

  // router logits
  k_router<<<(NBATCH * SEQ) / 8, 256, 0, stream>>>(x, router_w, logits);

  for (int b = 0; b < NBATCH; b++) {
    k_topk<<<1, 1024, 0, stream>>>(logits, b, sel, rw);
    k_gather_ln1<<<KSEL, 256, 0, stream>>>(x, sel, ln1_g, ln1_b, fx, h_bf, b);
    // qkv = h @ wqkv -> bf16
    k_gemm<1><<<dim3(3 * DMODEL / 128, KSEL / 128), 256, 0, stream>>>(
        h_bf, wqkv_t, KSEL, 3 * DMODEL, DMODEL, nullptr, qkv_bf, nullptr);
    // attention
    k_flash<<<NHEAD * (KSEL / 128), 256, 0, stream>>>(qkv_bf, o_bf);
    // x1 = fx + o @ wo
    k_gemm<2><<<dim3(DMODEL / 128, KSEL / 128), 256, 0, stream>>>(
        o_bf, wo_t, KSEL, DMODEL, DMODEL, x1, nullptr, fx);
    // h2 = LN2(x1)
    k_ln2<<<KSEL, 256, 0, stream>>>(x1, ln2_g, ln2_b, h2_bf);
    // g = gelu(h2 @ w1) -> bf16
    k_gemm<3><<<dim3(DFF / 128, KSEL / 128), 256, 0, stream>>>(
        h2_bf, w1_t, KSEL, DFF, DMODEL, nullptr, g_bf, nullptr);
    // xo = x1 + g @ w2
    k_gemm<2><<<dim3(DMODEL / 128, KSEL / 128), 256, 0, stream>>>(
        g_bf, w2_t, KSEL, DMODEL, DFF, xo, nullptr, x1);
    // out[b, sel] += rw * xo
    k_scatter<<<KSEL, 256, 0, stream>>>(out, sel, rw, xo, b);
  }
}
